// MeanAggregator_83872121356298
// MI455X (gfx1250) — matrix operand order repaired and checked by reading, not compiled, not run on hardware
//
#include <hip/hip_runtime.h>
#include <hip/hip_bf16.h>

#define UDIM 200000
#define FDIM 256
#define EDIM 128
#define BROWS 20000
#define KNBR 33
#define BN_EPS 1e-5f
#define NBLK ((UDIM + 127) / 128)   // 1563

typedef _Float16 v4h  __attribute__((ext_vector_type(4)));
typedef _Float16 v8h  __attribute__((ext_vector_type(8)));
typedef _Float16 v16h __attribute__((ext_vector_type(16)));
typedef float    v8f  __attribute__((ext_vector_type(8)));

union HFrag { v16h v; v8h h[2]; };

#if defined(__AMDGCN__) && __has_builtin(__builtin_amdgcn_global_load_async_to_lds_b128)
#define MG_ASYNC 1
typedef int mg_v4i __attribute__((vector_size(16)));
typedef __attribute__((address_space(1))) mg_v4i* mg_gptr;
typedef __attribute__((address_space(3))) mg_v4i* mg_lptr;
#else
#define MG_ASYNC 0
#endif

static __device__ __forceinline__ void mg_wait_async() {
#if __has_builtin(__builtin_amdgcn_s_wait_asynccnt)
  __builtin_amdgcn_s_wait_asynccnt(0);
#else
  asm volatile("s_wait_asynccnt 0x0" ::: "memory");
#endif
}

static __device__ __forceinline__ float fast_tanh(float x) {
#if __has_builtin(__builtin_amdgcn_tanhf)
  return __builtin_amdgcn_tanhf(x);
#else
  return tanhf(x);
#endif
}

// ---- Kernel 0: W [F,E] fp32 -> Wt [E][F] f16 (transposed for contiguous B frags)
__global__ __launch_bounds__(256) void mg_wt(const float* __restrict__ W,
                                             _Float16* __restrict__ Wt) {
  const int e = blockIdx.x;      // 0..127
  const int f = threadIdx.x;     // 0..255
  Wt[e * FDIM + f] = (_Float16)W[f * EDIM + e];
}

// ---- Kernel 1: h = features @ W + b (f16 WMMA), store h as f16, emit BN partials
__global__ __launch_bounds__(256) void mg_gemm(const float* __restrict__ feat,
                                               const _Float16* __restrict__ Wt,
                                               const float* __restrict__ bias,
                                               _Float16* __restrict__ hbuf,
                                               float* __restrict__ partSum,
                                               float* __restrict__ partSq) {
  __shared__ __align__(16) _Float16 smem[2 * 128 * 72];   // 36 KB
  _Float16 (*As)[72] = reinterpret_cast<_Float16(*)[72]>(smem);
  _Float16 (*Bs)[72] = reinterpret_cast<_Float16(*)[72]>(smem + 128 * 72);

  const int t    = threadIdx.x;
  const int lane = t & 31;
  const int wid  = t >> 5;
  const int wm   = (wid & 3) * 32;    // wave row offset (4 waves along M)
  const int wn   = (wid >> 2) * 64;   // wave col offset (2 waves along N)
  const long blockRow = (long)blockIdx.x * 128;
  const bool fullBlk  = (blockRow + 128 <= (long)UDIM);

  // loader coordinates
  const int c4 = t & 15;   // A: float4 column within 64-wide chunk
  const int r0 = t >> 4;   // A: base row
  const int c8 = t & 7;    // B: v8h column within chunk
  const int n0 = t >> 3;   // B: base row (E dim)

  v8f acc[2][4];
  for (int mt = 0; mt < 2; ++mt)
    for (int nt = 0; nt < 4; ++nt)
      for (int r = 0; r < 8; ++r) acc[mt][nt][r] = 0.f;

  // ---- software-pipelined A chunk prefetch (registers) ----
  float4 aReg[8];
  {
    const float* base = feat + blockRow * FDIM + c4 * 4;
    if (fullBlk) {
#pragma unroll
      for (int i = 0; i < 8; ++i)
        aReg[i] = *reinterpret_cast<const float4*>(base + (size_t)(r0 + i * 16) * FDIM);
    } else {
#pragma unroll
      for (int i = 0; i < 8; ++i) {
        const long gr = blockRow + r0 + i * 16;
        aReg[i] = make_float4(0.f, 0.f, 0.f, 0.f);
        if (gr < UDIM)
          aReg[i] = *reinterpret_cast<const float4*>(feat + gr * FDIM + c4 * 4);
      }
    }
  }

  for (int kc = 0; kc < FDIM; kc += 64) {
    __syncthreads();   // previous compute done reading LDS

    // ---- B chunk: async global->LDS (no conversion needed; ASYNCcnt path) ----
#pragma unroll
    for (int i = 0; i < 4; ++i) {
      const int n = n0 + i * 32;
#if MG_ASYNC
      __builtin_amdgcn_global_load_async_to_lds_b128(
          (mg_gptr)(Wt + n * FDIM + kc + c8 * 8),
          (mg_lptr)(&Bs[n][c8 * 8]), 0, 0);
#else
      v8h v = *reinterpret_cast<const v8h*>(Wt + n * FDIM + kc + c8 * 8);
      *reinterpret_cast<v8h*>(&Bs[n][c8 * 8]) = v;
#endif
    }

    // ---- convert + store the prefetched A registers ----
#pragma unroll
    for (int i = 0; i < 8; ++i) {
      v4h o;
      o[0] = (_Float16)aReg[i].x; o[1] = (_Float16)aReg[i].y;
      o[2] = (_Float16)aReg[i].z; o[3] = (_Float16)aReg[i].w;
      *reinterpret_cast<v4h*>(&As[r0 + i * 16][c4 * 4]) = o;
    }

    // ---- prefetch next A chunk (overlaps with this chunk's WMMAs) ----
    if (kc + 64 < FDIM) {
      const float* base = feat + blockRow * FDIM + (kc + 64) + c4 * 4;
      if (fullBlk) {
#pragma unroll
        for (int i = 0; i < 8; ++i)
          aReg[i] = *reinterpret_cast<const float4*>(base + (size_t)(r0 + i * 16) * FDIM);
      } else {
#pragma unroll
        for (int i = 0; i < 8; ++i) {
          const long gr = blockRow + r0 + i * 16;
          aReg[i] = make_float4(0.f, 0.f, 0.f, 0.f);
          if (gr < UDIM)
            aReg[i] = *reinterpret_cast<const float4*>(feat + gr * FDIM + (kc + 64) + c4 * 4);
        }
      }
    }

#if MG_ASYNC
    mg_wait_async();   // this wave's async B stores to LDS complete
#endif
    __syncthreads();

    // ---- 16 WMMAs per wave per chunk ----
    for (int ks = 0; ks < 2; ++ks) {
      const int kbA = ks * 32 + (lane >> 4) * 8;   // A: K 0-7/16-23 per half-wave
      const int kbB = ks * 32 + (lane >> 4) * 8;   // B: K 0-7/16-23 per half-wave
      HFrag a[2], b[4];
#pragma unroll
      for (int mt = 0; mt < 2; ++mt) {
        const int row = wm + mt * 16 + (lane & 15);
        a[mt].h[0] = *reinterpret_cast<const v8h*>(&As[row][kbA]);
        a[mt].h[1] = *reinterpret_cast<const v8h*>(&As[row][kbA + 16]);
      }
#pragma unroll
      for (int nt = 0; nt < 4; ++nt) {
        const int col = wn + nt * 16 + (lane & 15);
        b[nt].h[0] = *reinterpret_cast<const v8h*>(&Bs[col][kbB]);
        b[nt].h[1] = *reinterpret_cast<const v8h*>(&Bs[col][kbB + 16]);
      }
#pragma unroll
      for (int mt = 0; mt < 2; ++mt)
#pragma unroll
        for (int nt = 0; nt < 4; ++nt)
          acc[mt][nt] = __builtin_amdgcn_wmma_f32_16x16x32_f16(
              false, a[mt].v, false, b[nt].v, (short)0, acc[mt][nt], false, false);
    }
  }

  __syncthreads();   // all fragment reads done; reuse LDS as f16 staging tile
  _Float16 (*stage)[136] = reinterpret_cast<_Float16(*)[136]>(smem);
  {
    const int rb = wm + 8 * (lane >> 4);
    const int nc = lane & 15;
#pragma unroll
    for (int nt = 0; nt < 4; ++nt) {
      const int n = wn + nt * 16 + nc;
      const float bn = bias[n];
#pragma unroll
      for (int mt = 0; mt < 2; ++mt)
#pragma unroll
        for (int r = 0; r < 8; ++r) {
          const int m = rb + mt * 16 + r;
          stage[m][n] = (_Float16)(acc[mt][nt][r] + bn);
        }
    }
  }
  __syncthreads();

  { // deterministic per-block column partials
    const int j = t & 127;
    float s = 0.f;
    if (t < 128) {
      if (fullBlk) {
        for (int m = 0; m < 128; ++m) s += (float)stage[m][j];
      } else {
        for (int m = 0; m < 128; ++m)
          if (blockRow + m < UDIM) s += (float)stage[m][j];
      }
      partSum[blockIdx.x * 128 + j] = s;
    } else {
      if (fullBlk) {
        for (int m = 0; m < 128; ++m) { const float v = (float)stage[m][j]; s += v * v; }
      } else {
        for (int m = 0; m < 128; ++m)
          if (blockRow + m < UDIM) { const float v = (float)stage[m][j]; s += v * v; }
      }
      partSq[blockIdx.x * 128 + j] = s;
    }
  }
  { // coalesced f16 store of the h tile
    const int row = t >> 1;
    const int h0  = (t & 1) * 64;
    const long gr = blockRow + row;
    if (gr < UDIM) {
#pragma unroll
      for (int i = 0; i < 8; ++i) {
        v8h v = *reinterpret_cast<const v8h*>(&stage[row][h0 + i * 8]);
        *reinterpret_cast<v8h*>(hbuf + gr * EDIM + h0 + i * 8) = v;
      }
    }
  }
}

// ---- Kernel 2: reduce partials -> fused scale/shift (single block, fixed order)
__global__ __launch_bounds__(128) void mg_stats(const float* __restrict__ partSum,
                                                const float* __restrict__ partSq,
                                                const float* __restrict__ gamma,
                                                const float* __restrict__ beta,
                                                float* __restrict__ scsh) {
  const int j = threadIdx.x;   // 0..127
  double s = 0.0, q = 0.0;
  for (int i = 0; i < NBLK; ++i) {
    s += (double)partSum[i * 128 + j];
    q += (double)partSq[i * 128 + j];
  }
  const float mu  = (float)(s / (double)UDIM);
  const float var = (float)(q / (double)UDIM) - mu * mu;
  const float sc  = gamma[j] * rsqrtf(var + BN_EPS);
  scsh[j]        = sc;
  scsh[EDIM + j] = beta[j] - mu * sc;
}

// ---- Kernel 3: gather + tanh-normalize + mean over K (one wave per output row)
__global__ __launch_bounds__(256) void mg_gather(const _Float16* __restrict__ hbuf,
                                                 const int* __restrict__ sidx,
                                                 const float* __restrict__ scsh,
                                                 float* __restrict__ out) {
  const int lane = threadIdx.x & 31;
  const int row  = blockIdx.x * 8 + (threadIdx.x >> 5);   // 2500*8 == 20000 exact
  const int e0   = lane * 4;

  float sc[4], sh[4];
#pragma unroll
  for (int j = 0; j < 4; ++j) { sc[j] = scsh[e0 + j]; sh[j] = scsh[EDIM + e0 + j]; }

  float a0 = 0.f, a1 = 0.f, a2 = 0.f, a3 = 0.f;
  const int* ip = sidx + row * KNBR;   // wave-uniform -> scalar loads
  for (int k = 0; k < KNBR; ++k) {
    const int id = ip[k];
    const v4h hv = *reinterpret_cast<const v4h*>(hbuf + (long)id * EDIM + e0);
    a0 += fast_tanh(sc[0] * (float)hv[0] + sh[0]);
    a1 += fast_tanh(sc[1] * (float)hv[1] + sh[1]);
    a2 += fast_tanh(sc[2] * (float)hv[2] + sh[2]);
    a3 += fast_tanh(sc[3] * (float)hv[3] + sh[3]);
  }
  const float inv = 1.f / (float)KNBR;
  float4 o = make_float4(a0 * inv, a1 * inv, a2 * inv, a3 * inv);
  *reinterpret_cast<float4*>(out + (long)row * EDIM + e0) = o;
}

extern "C" void kernel_launch(void* const* d_in, const int* in_sizes, int n_in,
                              void* d_out, int out_size, void* d_ws, size_t ws_size,
                              hipStream_t stream) {
  (void)in_sizes; (void)n_in; (void)out_size; (void)ws_size;

  const float* feat  = (const float*)d_in[0];   // [U,F]
  const float* W     = (const float*)d_in[1];   // [F,E]
  const float* bias  = (const float*)d_in[2];   // [E]
  const float* gamma = (const float*)d_in[3];   // [E]
  const float* beta  = (const float*)d_in[4];   // [E]
  const int*   sidx  = (const int*)d_in[5];     // [B,K] int32
  float*       out   = (float*)d_out;           // [B,E]

  // workspace layout (all 16B aligned)
  char* ws = (char*)d_ws;
  const size_t hbytes  = (size_t)UDIM * EDIM * sizeof(_Float16);   // 51,200,000
  const size_t wtbytes = (size_t)EDIM * FDIM * sizeof(_Float16);   // 65,536
  const size_t pbytes  = (size_t)NBLK * 128 * sizeof(float);       // 800,256
  _Float16* hbuf    = (_Float16*)ws;
  _Float16* Wt      = (_Float16*)(ws + hbytes);
  float*    partSum = (float*)(ws + hbytes + wtbytes);
  float*    partSq  = (float*)(ws + hbytes + wtbytes + pbytes);
  float*    scsh    = (float*)(ws + hbytes + wtbytes + 2 * pbytes);

  mg_wt<<<EDIM, FDIM, 0, stream>>>(W, Wt);
  mg_gemm<<<NBLK, 256, 0, stream>>>(feat, Wt, bias, hbuf, partSum, partSq);
  mg_stats<<<1, 128, 0, stream>>>(partSum, partSq, gamma, beta, scsh);
  mg_gather<<<BROWS / 8, 256, 0, stream>>>(hbuf, sidx, scsh, out);
}
